// FP8GroupLinear_5050881540804
// MI455X (gfx1250) — 8-bit matrix operand order repaired and checked by reading, compiled, not run on hardware
//
#include <hip/hip_runtime.h>

typedef __attribute__((ext_vector_type(16))) int   v16i;
typedef __attribute__((ext_vector_type(8)))  float v8f;

#define FP8_MAX 448.0f
#define EPSQ 1e-4f

__device__ __forceinline__ unsigned int f32_to_e4m3(float f) {
    unsigned int u = __float_as_uint(f);
    unsigned int s = (u >> 24) & 0x80u;
    u &= 0x7FFFFFFFu;
    if (u > 0x43E00000u) u = 0x43E00000u;
    if (u < 0x3C800000u) {
        float d = __uint_as_float(u) * 512.0f;
        int q = __float2int_rn(d);
        return s | (unsigned int)q;
    }
    unsigned int r  = u + 0x0007FFFFu + ((u >> 20) & 1u);
    int          e4 = (int)(r >> 23) - 127 + 7;
    unsigned int m  = (r >> 20) & 0x7u;
    if (e4 >= 16) { e4 = 15; m = 6; }
    return s | ((unsigned int)e4 << 3) | m;
}

__global__ void qx_kernel(const float* __restrict__ x,
                          unsigned char* __restrict__ xq,
                          float* __restrict__ xs_t, int nblocks, int nbk, int M) {
    int wave = (blockIdx.x * blockDim.x + threadIdx.x) >> 5;
    int lane = threadIdx.x & 31;
    if (wave >= nblocks) return;
    const float4 v = *(const float4*)(x + (size_t)wave * 128 + lane * 4);
    float a = fmaxf(fmaxf(fabsf(v.x), fabsf(v.y)), fmaxf(fabsf(v.z), fabsf(v.w)));
    #pragma unroll
    for (int off = 16; off >= 1; off >>= 1)
        a = fmaxf(a, __shfl_xor(a, off, 32));
    float amax = fmaxf(a, EPSQ);
    float inv  = FP8_MAX / amax;
    unsigned int pk = f32_to_e4m3(v.x * inv)
                    | (f32_to_e4m3(v.y * inv) << 8)
                    | (f32_to_e4m3(v.z * inv) << 16)
                    | (f32_to_e4m3(v.w * inv) << 24);
    *(unsigned int*)(xq + (size_t)wave * 128 + lane * 4) = pk;
    if (lane == 0) {
        int m  = wave / nbk;
        int kb = wave - m * nbk;
        xs_t[(size_t)kb * M + m] = amax * (1.0f / FP8_MAX);
    }
}

__global__ void qw_kernel(const float* __restrict__ w,
                          unsigned char* __restrict__ wq,
                          float* __restrict__ wsc, int K, int nbk) {
    int bid = blockIdx.x;
    int kb  = bid % nbk;
    int nbg = bid / nbk;
    const float* base = w + (size_t)nbg * 128 * K + (size_t)kb * 128;
    int tid = threadIdx.x;
    int row = tid >> 1;
    int col = (tid & 1) * 64;
    const float4* p = (const float4*)(base + (size_t)row * K + col);
    float4 vals[16];
    float a = 0.0f;
    #pragma unroll
    for (int i = 0; i < 16; ++i) {
        vals[i] = p[i];
        a = fmaxf(a, fmaxf(fmaxf(fabsf(vals[i].x), fabsf(vals[i].y)),
                           fmaxf(fabsf(vals[i].z), fabsf(vals[i].w))));
    }
    #pragma unroll
    for (int off = 16; off >= 1; off >>= 1)
        a = fmaxf(a, __shfl_xor(a, off, 32));
    __shared__ float red[8];
    if ((tid & 31) == 0) red[tid >> 5] = a;
    __syncthreads();
    if (tid == 0) {
        float b = EPSQ;
        #pragma unroll
        for (int i = 0; i < 8; ++i) b = fmaxf(b, red[i]);
        red[0] = b;
        wsc[bid] = b * (1.0f / FP8_MAX);
    }
    __syncthreads();
    float inv = FP8_MAX / red[0];
    unsigned int* q = (unsigned int*)(wq + (size_t)nbg * 128 * K + (size_t)kb * 128
                                         + (size_t)row * K + col);
    #pragma unroll
    for (int i = 0; i < 16; ++i) {
        q[i] = f32_to_e4m3(vals[i].x * inv)
             | (f32_to_e4m3(vals[i].y * inv) << 8)
             | (f32_to_e4m3(vals[i].z * inv) << 16)
             | (f32_to_e4m3(vals[i].w * inv) << 24);
    }
}

__global__ __launch_bounds__(256)
void gemm_kernel(const unsigned char* __restrict__ xq, const float* __restrict__ xs_t,
                 const unsigned char* __restrict__ wq, const float* __restrict__ wsc,
                 const int* __restrict__ gidx, float* __restrict__ out,
                 int M, int N, int K) {
    __shared__ unsigned char ldsA[64 * 128];
    __shared__ unsigned char ldsB[256 * 128];

    const int tid   = threadIdx.x;
    const int lane  = tid & 31;
    const int wid   = tid >> 5;
    const int wm    = wid >> 2;
    const int wn    = wid & 3;
    const int mWG   = blockIdx.y * 64;
    const int nWG   = blockIdx.x * 256;
    const int mbase = mWG + wm * 32;
    const int nbase = nWG + wn * 64;
    if (mbase >= M || nbase >= N) return;
    const int nbk = K >> 7;
    const int nbn = N >> 7;
    const int g   = gidx[mWG];
    const unsigned char* wg = wq + (size_t)g * N * K;

    const int lrow = lane & 15;
    const int lg   = lane >> 4;

    const int arow_s = tid >> 2;
    const int acol_s = (tid & 3) * 32;
    const unsigned char* agbase = xq + (size_t)(mWG + arow_s) * K + acol_s;
    unsigned char*       albase = &ldsA[arow_s * 128 + acol_s];
    const int brow_s = tid >> 3;
    const int bcol_s = (tid & 7) * 16;
    const unsigned char* bgbase = wg + (size_t)(nWG + brow_s) * K + bcol_s;
    unsigned char*       blbase = &ldsB[brow_s * 128 + bcol_s];
    const size_t bgstride = (size_t)32 * K;

    const unsigned char* afrag[2];
    #pragma unroll
    for (int aI = 0; aI < 2; ++aI)
        afrag[aI] = &ldsA[(wm * 32 + aI * 16 + lrow) * 128];
    const unsigned char* bfrag[4];
    #pragma unroll
    for (int t = 0; t < 4; ++t)
        bfrag[t] = &ldsB[(wn * 64 + t * 16 + lrow) * 128];

    const int    wsbase = (g * nbn + (nbase >> 7)) * nbk;
    const float* xrow   = xs_t + mbase + lg * 8;

    v8f acc[2][4];
    #pragma unroll
    for (int aI = 0; aI < 2; ++aI)
        #pragma unroll
        for (int t = 0; t < 4; ++t)
            #pragma unroll
            for (int r = 0; r < 8; ++r) acc[aI][t][r] = 0.0f;

    v8f z;
    #pragma unroll
    for (int r = 0; r < 8; ++r) z[r] = 0.0f;

    for (int ks = 0; ks < nbk; ++ks) {
        const unsigned char* ag = agbase + ks * 128;
        int4 a0 = *(const int4*)(ag);
        int4 a1 = *(const int4*)(ag + 16);
        int4 bt[8];
        #pragma unroll
        for (int i = 0; i < 8; ++i)
            bt[i] = *(const int4*)(bgbase + ks * 128 + (size_t)i * bgstride);

        __syncthreads();
        *(int4*)(albase)      = a0;
        *(int4*)(albase + 16) = a1;
        #pragma unroll
        for (int i = 0; i < 8; ++i)
            *(int4*)(blbase + i * 4096) = bt[i];
        __syncthreads();

        v16i A[2];
        #pragma unroll
        for (int aI = 0; aI < 2; ++aI)
            #pragma unroll
            for (int h = 0; h < 2; ++h)
                #pragma unroll
                for (int p = 0; p < 4; ++p) {
                    int2 d = *(const int2*)(afrag[aI] + h * 64 + (p >> 1) * 32 + lg * 16 + (p & 1) * 8);
                    A[aI][h * 8 + p * 2 + 0] = d.x;
                    A[aI][h * 8 + p * 2 + 1] = d.y;
                }

        const float  sw  = wsc[wsbase + ks];
        const float* xsk = xrow + (size_t)ks * M;
        float4 s0 = *(const float4*)(xsk);
        float4 s1 = *(const float4*)(xsk + 4);
        float4 s2 = *(const float4*)(xsk + 16);
        float4 s3 = *(const float4*)(xsk + 20);
        float sc[2][8] = {
            { s0.x * sw, s0.y * sw, s0.z * sw, s0.w * sw,
              s1.x * sw, s1.y * sw, s1.z * sw, s1.w * sw },
            { s2.x * sw, s2.y * sw, s2.z * sw, s2.w * sw,
              s3.x * sw, s3.y * sw, s3.z * sw, s3.w * sw } };

        #pragma unroll
        for (int t = 0; t < 4; ++t) {
            v16i Bt;
            #pragma unroll
            for (int q4 = 0; q4 < 4; ++q4) {
                int4 d = *(const int4*)(bfrag[t] + q4 * 32 + lg * 16);
                Bt[q4 * 4 + 0] = d.x;  Bt[q4 * 4 + 1] = d.y;
                Bt[q4 * 4 + 2] = d.z;  Bt[q4 * 4 + 3] = d.w;
            }
            #pragma unroll
            for (int aI = 0; aI < 2; ++aI) {
                v8f part = __builtin_amdgcn_wmma_f32_16x16x128_fp8_fp8(
                    A[aI], Bt, (short)0, z, false, false);
                #pragma unroll
                for (int r = 0; r < 8; ++r)
                    acc[aI][t][r] = fmaf(part[r], sc[aI][r], acc[aI][t][r]);
            }
        }
    }

    #pragma unroll
    for (int aI = 0; aI < 2; ++aI)
        #pragma unroll
        for (int t = 0; t < 4; ++t) {
            float* o = out + (size_t)(mbase + aI * 16 + lg * 8) * N
                           + nbase + t * 16 + lrow;
            #pragma unroll
            for (int r = 0; r < 8; ++r)
                o[(size_t)r * N] = acc[aI][t][r];
        }
}

extern "C" void kernel_launch(void* const* d_in, const int* in_sizes, int n_in,
                              void* d_out, int out_size, void* d_ws, size_t ws_size,
                              hipStream_t stream) {
    const float* x    = (const float*)d_in[0];
    const float* w    = (const float*)d_in[1];
    const int*   gidx = (const int*)d_in[3];
    float*       out  = (float*)d_out;

    const int M = in_sizes[3];
    const int K = in_sizes[0] / M;
    const int G = in_sizes[2];
    const int N = (int)((long long)in_sizes[1] / ((long long)G * K));
    const int nbk = K / 128, nbn = N / 128;

    unsigned char* xq  = (unsigned char*)d_ws;
    unsigned char* wqp = xq + (size_t)M * K;
    float*         xsp = (float*)(wqp + (size_t)G * N * K);
    float*         wsc = xsp + (size_t)M * nbk;

    {
        int nblocks = M * nbk;
        int blocks  = (nblocks * 32 + 255) / 256;
        qx_kernel<<<blocks, 256, 0, stream>>>(x, xq, xsp, nblocks, nbk, M);
    }
    qw_kernel<<<G * nbn * nbk, 256, 0, stream>>>(w, wqp, wsc, K, nbk);

    dim3 grid((N + 255) / 256, (M + 63) / 64);
    gemm_kernel<<<grid, 256, 0, stream>>>(xq, xsp, wqp, wsc, gidx, out, M, N, K);
}
